// SPMTrack_Inference_DINOv2_47210280518288
// MI455X (gfx1250) — matrix operand order repaired and checked by reading, not compiled, not run on hardware
//
#include <hip/hip_runtime.h>
#include <hip/hip_bf16.h>

typedef __attribute__((ext_vector_type(16))) _Float16 v16h;
typedef __attribute__((ext_vector_type(8)))  _Float16 v8h;
typedef __attribute__((ext_vector_type(8)))  float    v8f;
typedef __attribute__((ext_vector_type(4)))  unsigned int v4u;
typedef __attribute__((ext_vector_type(8)))  int      v8i;
typedef __attribute__((ext_vector_type(4)))  int      v4i;

#define T_TOK 449
#define BATCH 16
#define DMODEL 768
#define NHEAD 12
#define HDIM 64
#define NLAYER 12
#define MTOK (BATCH * T_TOK)   // 7184
#define LORA_K 64              // E*R = 4*16
#define SCALE_LORA 1.0f        // 16 / R
#define LDST 40                // B-tile LDS row stride (halves), padded vs 32

// ---------------------------------------------------------------------------
// WMMA helpers (gfx1250, wave32, v_wmma_f32_16x16x32_f16)
// ---------------------------------------------------------------------------
__device__ __forceinline__ v8f wmma_f32(v16h a, v16h b, v8f c) {
  return __builtin_amdgcn_wmma_f32_16x16x32_f16(false, a, false, b, (short)0, c,
                                                false, false);
}

// ---------------------------------------------------------------------------
// TDM: load a 2D fp32 tile (32 wide x 64 rows) from global into LDS.
// D# packing per CDNA5 ISA:
//   group0: count=1 | lds_addr | global_addr[56:0] | type=2
//   group1: data_size=4B, tensor_dim0=remK, tensor_dim1=remM,
//           tile_dim0=32, tile_dim1=64, tensor_dim0_stride=lda
// OOB reads (beyond remK/remM) return zero -> hardware edge masking.
// ---------------------------------------------------------------------------
__device__ __forceinline__ void tdm_load_2d_f32(const float* g,
                                                unsigned int lds_byte, int remK,
                                                int remM, int lda) {
  unsigned long long ga = (unsigned long long)g;
  v4u g0;
  g0[0] = 1u;                                   // count=1, user descriptor
  g0[1] = lds_byte;                             // lds_addr (bytes)
  g0[2] = (unsigned int)ga;                     // global_addr[31:0]
  g0[3] = (unsigned int)((ga >> 32) & 0x1FFFFFFull) | (2u << 30);  // type=2
  unsigned int rk = (unsigned int)remK, rm = (unsigned int)remM;
  unsigned long long st = (unsigned long long)(unsigned int)lda;
  v8i g1;
  g1[0] = (int)(2u << 16);                                      // data_size=4B
  g1[1] = (int)((rk & 0xFFFFu) << 16);                          // dim0 lo
  g1[2] = (int)(((rk >> 16) & 0xFFFFu) | ((rm & 0xFFFFu) << 16));
  g1[3] = (int)(((rm >> 16) & 0xFFFFu) | (32u << 16));          // tile_dim0=32
  g1[4] = (int)64u;                              // tile_dim1=64, tile_dim2=0
  g1[5] = (int)(unsigned int)st;                 // stride0[31:0]
  g1[6] = (int)(unsigned int)((st >> 32) & 0xFFFFu);  // stride0[47:32]
  g1[7] = 0;
  v4i z4 = {};
#if defined(__clang_major__) && __clang_major__ >= 23
  v8i z8 = {};
  __builtin_amdgcn_tensor_load_to_lds(g0, g1, z4, z4, z8, 0);
#else
  __builtin_amdgcn_tensor_load_to_lds(g0, g1, z4, z4, 0);
#endif
}

// A fragment (16x32 f16) from fp32 LDS tile written by TDM (row stride 32).
// ISA layout: row = lane&15; k = 8*(lane>>4) + (j&7) + 16*(j>>3)
__device__ __forceinline__ v16h frag_a_lds32(const float* src) {
  int lane = threadIdx.x & 31;
  const float* p = src + (lane & 15) * 32 + ((lane >> 4) << 3);
  float4 x0 = ((const float4*)p)[0];
  float4 x1 = ((const float4*)p)[1];
  float4 y0 = ((const float4*)(p + 16))[0];
  float4 y1 = ((const float4*)(p + 16))[1];
  float va[16] = {x0.x, x0.y, x0.z, x0.w, x1.x, x1.y, x1.z, x1.w,
                  y0.x, y0.y, y0.z, y0.w, y1.x, y1.y, y1.z, y1.w};
  v16h f;
#pragma unroll
  for (int j = 0; j < 16; ++j) f[j] = (_Float16)va[j];
  return f;
}

// A fragment (16x32 f16) from row-major f16 LDS tile (ld in halves).
__device__ __forceinline__ v16h frag_a_lds(const _Float16* src, int ld) {
  int lane = threadIdx.x & 31;
  const _Float16* p = src + (lane & 15) * ld + ((lane >> 4) << 4);
  v8h lo = *(const v8h*)p;
  v8h hi = *(const v8h*)(p + 8);
  v16h f;
#pragma unroll
  for (int j = 0; j < 8; ++j) { f[j] = lo[j]; f[j + 8] = hi[j]; }
  return f;
}

// B fragment (32x16 f16) from TRANSPOSED LDS tile lB[n][k] (stride LDST).
// ISA layout: col = lane&15; k = 8*(lane>>4) + (j&7) + 16*(j>>3).
__device__ __forceinline__ v16h frag_b_ldsT(const _Float16* srcT) {
  int lane = threadIdx.x & 31;
  const _Float16* p = srcT + (lane & 15) * LDST + ((lane >> 4) << 3);
  v8h lo = *(const v8h*)p;
  v8h hi = *(const v8h*)(p + 16);
  v16h f;
#pragma unroll
  for (int j = 0; j < 8; ++j) { f[j] = lo[j]; f[j + 8] = hi[j]; }
  return f;
}

// A fragment (16x32 f16) straight from row-major (M x K) fp32 global memory.
// Used by attention (Q). Branchless: clamp row, vector loads, select 0.
__device__ __forceinline__ v16h frag_a_rm(const float* src, int ld, int row0,
                                          int k0, int rlim, int klim) {
  int lane = threadIdx.x & 31;
  int r0 = row0 + (lane & 15);
  bool rok = r0 < rlim;
  int r = rok ? r0 : (rlim - 1);
  int kb = k0 + ((lane >> 4) << 3);
  const float* p = src + (size_t)r * ld;
  v16h f;
  if (k0 + 32 <= klim) {
    const float4* q0 = (const float4*)(p + kb);
    const float4* q1 = (const float4*)(p + kb + 16);
    float4 a0 = q0[0], a1 = q0[1], b0 = q1[0], b1 = q1[1];
    float va[16] = {a0.x, a0.y, a0.z, a0.w, a1.x, a1.y, a1.z, a1.w,
                    b0.x, b0.y, b0.z, b0.w, b1.x, b1.y, b1.z, b1.w};
#pragma unroll
    for (int j = 0; j < 16; ++j) f[j] = (_Float16)(rok ? va[j] : 0.f);
  } else {
#pragma unroll
    for (int j = 0; j < 16; ++j) {
      int k = kb + (j & 7) + ((j >> 3) << 4);
      bool kok = k < klim;
      float v = p[kok ? k : (klim - 1)];
      f[j] = (_Float16)((rok && kok) ? v : 0.f);
    }
  }
  return f;
}

// B fragment (32x16 f16) from row-major (K x N) fp32 global (attention V).
__device__ __forceinline__ v16h frag_b_kn(const float* src, int ld, int k0,
                                          int n0, int klim, int nlim) {
  int lane = threadIdx.x & 31;
  int n1 = n0 + (lane & 15);
  bool nok = n1 < nlim;
  int n = nok ? n1 : (nlim - 1);
  int kb = k0 + ((lane >> 4) << 4);
  v16h f;
  if (k0 + 32 <= klim) {
#pragma unroll
    for (int j = 0; j < 16; ++j) {
      float v = src[(size_t)(kb + j) * ld + n];
      f[j] = (_Float16)(nok ? v : 0.f);
    }
  } else {
#pragma unroll
    for (int j = 0; j < 16; ++j) {
      int k = kb + j;
      bool kok = k < klim;
      float v = src[(size_t)(kok ? k : (klim - 1)) * ld + n];
      f[j] = (_Float16)((nok && kok) ? v : 0.f);
    }
  }
  return f;
}

// B fragment from row-major (N x K) fp32 global (attention K as B=K^T).
__device__ __forceinline__ v16h frag_b_nk(const float* src, int ld, int n0,
                                          int k0, int nlim) {
  int lane = threadIdx.x & 31;
  int n1 = n0 + (lane & 15);
  bool nok = n1 < nlim;
  int n = nok ? n1 : (nlim - 1);
  int kb = k0 + ((lane >> 4) << 3);
  const float* p = src + (size_t)n * ld + kb;
  const float4* q = (const float4*)p;
  const float4* q2 = (const float4*)(p + 16);
  float4 a0 = q[0], a1 = q[1], a2 = q2[0], a3 = q2[1];
  float va[16] = {a0.x, a0.y, a0.z, a0.w, a1.x, a1.y, a1.z, a1.w,
                  a2.x, a2.y, a2.z, a2.w, a3.x, a3.y, a3.z, a3.w};
  v16h f;
#pragma unroll
  for (int j = 0; j < 16; ++j) f[j] = (_Float16)(nok ? va[j] : 0.f);
  return f;
}

__device__ __forceinline__ float gelu_f(float x) {
  float x3 = x * x * x;
  return 0.5f * x * (1.f + tanhf(0.7978845608028654f * (x + 0.044715f * x3)));
}

// B tile staging: W tile (32 k x 64 n) fp32 -> f16 LDS, stored transposed
// lB[n][k]. 256 threads; thread -> (n = t%64, kq = (t/64)*8).
template <bool EDGE>
__device__ __forceinline__ void stage_b_f16(const float* __restrict__ W,
                                            int ldw, int K, int k0,
                                            _Float16* lB) {
  int t = threadIdx.x;
  int n = t & 63, kq = (t >> 6) << 3;
  v8h hv;
  if (!EDGE) {
#pragma unroll
    for (int i = 0; i < 8; ++i)
      hv[i] = (_Float16)W[(size_t)(k0 + kq + i) * ldw + n];
    if (k0 + 64 <= K)
      __builtin_prefetch(W + (size_t)(k0 + 32 + kq) * ldw + n, 0, 1);
  } else {
#pragma unroll
    for (int i = 0; i < 8; ++i) {
      int kk = k0 + kq + i;
      bool kok = kk < K;
      float v = W[(size_t)(kok ? kk : (K - 1)) * ldw + n];
      hv[i] = (_Float16)(kok ? v : 0.f);
    }
  }
  *(v8h*)(lB + n * LDST + kq) = hv;
}

// ---------------------------------------------------------------------------
// Fused GEMM: C = A1(MxK1)*W1(K1xN) + s2 * A2(MxK2)*W2(K2xN) + bias, opt GELU.
// WG = 8 waves, macro 64x64. A tile staged by the Tensor Data Mover (fp32 in
// LDS, hardware zero-fill at M/K edges); B tile staged transposed as f16.
// Requires: N multiple of 64 (true for all call sites); K2 multiple of 32.
// ---------------------------------------------------------------------------
__global__ __launch_bounds__(256) void gemm_wmma(
    const float* __restrict__ A1, int lda1, const float* __restrict__ W1,
    int ldw1, int K1, const float* __restrict__ A2, int lda2,
    const float* __restrict__ W2, int ldw2, int K2, float s2,
    const float* __restrict__ bias, float* __restrict__ C, int ldc, int M,
    int N, int act) {
  __shared__ float    lA[64 * 32];
  __shared__ _Float16 lB[64 * LDST];
  int wave = threadIdx.x >> 5;
  int wr = (wave & 3) * 16;   // wave row offset in macro tile
  int wc = (wave >> 2) * 32;  // wave col offset in macro tile
  int rowblk = blockIdx.x * 64;
  int colblk = blockIdx.y * 64;
  unsigned int ldsA = (unsigned int)(unsigned long long)&lA[0];
  v8f acc0 = {}, acc1 = {};

  const float* Abase = A1 + (size_t)rowblk * lda1;
  int Kmain = K1 & ~31;
  for (int k0 = 0; k0 < Kmain; k0 += 32) {
    __syncthreads();
    if (wave == 0)
      tdm_load_2d_f32(Abase + k0, ldsA, K1 - k0, M - rowblk, lda1);
    stage_b_f16<false>(W1 + colblk, ldw1, K1, k0, lB);
    if (wave == 0) __builtin_amdgcn_s_wait_tensorcnt(0);
    __syncthreads();
    v16h a = frag_a_lds32(lA + wr * 32);
    v16h b0 = frag_b_ldsT(lB + wc * LDST);
    v16h b1 = frag_b_ldsT(lB + (wc + 16) * LDST);
    acc0 = wmma_f32(a, b0, acc0);
    acc1 = wmma_f32(a, b1, acc1);
  }
  if (Kmain < K1) {  // K edge tile (patch embed K=588): TDM zero-fills K
    __syncthreads();
    if (wave == 0)
      tdm_load_2d_f32(Abase + Kmain, ldsA, K1 - Kmain, M - rowblk, lda1);
    stage_b_f16<true>(W1 + colblk, ldw1, K1, Kmain, lB);
    if (wave == 0) __builtin_amdgcn_s_wait_tensorcnt(0);
    __syncthreads();
    v16h a = frag_a_lds32(lA + wr * 32);
    v16h b0 = frag_b_ldsT(lB + wc * LDST);
    v16h b1 = frag_b_ldsT(lB + (wc + 16) * LDST);
    acc0 = wmma_f32(a, b0, acc0);
    acc1 = wmma_f32(a, b1, acc1);
  }

  if (K2 > 0) {  // LoRA segment, K2 = 64 (multiple of 32)
    v8f t0 = {}, t1 = {};
    const float* A2base = A2 + (size_t)rowblk * lda2;
    for (int k0 = 0; k0 < K2; k0 += 32) {
      __syncthreads();
      if (wave == 0)
        tdm_load_2d_f32(A2base + k0, ldsA, K2 - k0, M - rowblk, lda2);
      stage_b_f16<false>(W2 + colblk, ldw2, K2, k0, lB);
      if (wave == 0) __builtin_amdgcn_s_wait_tensorcnt(0);
      __syncthreads();
      v16h a = frag_a_lds32(lA + wr * 32);
      v16h b0 = frag_b_ldsT(lB + wc * LDST);
      v16h b1 = frag_b_ldsT(lB + (wc + 16) * LDST);
      t0 = wmma_f32(a, b0, t0);
      t1 = wmma_f32(a, b1, t1);
    }
#pragma unroll
    for (int i = 0; i < 8; ++i) {
      acc0[i] += s2 * t0[i];
      acc1[i] += s2 * t1[i];
    }
  }

  int lane = threadIdx.x & 31;
  int n0 = colblk + wc + (lane & 15);
  int rb = rowblk + wr + ((lane >> 4) << 3);
  float bv0 = bias ? bias[n0] : 0.f;
  float bv1 = bias ? bias[n0 + 16] : 0.f;
#pragma unroll
  for (int i = 0; i < 8; ++i) {
    int m = rb + i;
    if (m < M) {
      float v0 = acc0[i] + bv0;
      float v1 = acc1[i] + bv1;
      if (act) { v0 = gelu_f(v0); v1 = gelu_f(v1); }
      C[(size_t)m * ldc + n0] = v0;
      C[(size_t)m * ldc + n0 + 16] = v1;
    }
  }
}

// ---------------------------------------------------------------------------
// Flash attention: one wave handles 16 query rows of one (batch, head).
// qkv layout: (MTOK, 2304) = [q | k | v], head h at columns h*64.
// ---------------------------------------------------------------------------
__global__ __launch_bounds__(32) void attn_wmma(const float* __restrict__ qkv,
                                                float* __restrict__ out) {
  __shared__ _Float16 pt[16 * 32];
  int bh = blockIdx.x;
  int b = bh / NHEAD, h = bh % NHEAD;
  int q0 = blockIdx.y * 16;
  const float* qptr = qkv + (size_t)b * T_TOK * 2304 + h * HDIM;
  const float* kptr = qptr + DMODEL;
  const float* vptr = qptr + 2 * DMODEL;
  int lane = threadIdx.x & 31;

  v16h qf0 = frag_a_rm(qptr, 2304, q0, 0, T_TOK, HDIM);
  v16h qf1 = frag_a_rm(qptr, 2304, q0, 32, T_TOK, HDIM);
  v8f o0 = {}, o1 = {}, o2 = {}, o3 = {};
  float mrow[8], lrow[8];
#pragma unroll
  for (int i = 0; i < 8; ++i) { mrow[i] = -1e30f; lrow[i] = 0.f; }

  for (int kt = 0; kt < T_TOK; kt += 32) {
    v8f s0 = {}, s1 = {};
    {
      v16h kfa = frag_b_nk(kptr, 2304, kt, 0, T_TOK);
      v16h kfb = frag_b_nk(kptr, 2304, kt, 32, T_TOK);
      s0 = wmma_f32(qf0, kfa, s0);
      s0 = wmma_f32(qf1, kfb, s0);
    }
    {
      v16h kfa = frag_b_nk(kptr, 2304, kt + 16, 0, T_TOK);
      v16h kfb = frag_b_nk(kptr, 2304, kt + 16, 32, T_TOK);
      s1 = wmma_f32(qf0, kfa, s1);
      s1 = wmma_f32(qf1, kfb, s1);
    }
    int c0 = kt + (lane & 15), c1 = kt + 16 + (lane & 15);
#pragma unroll
    for (int i = 0; i < 8; ++i) {
      s0[i] = (c0 < T_TOK) ? s0[i] * 0.125f : -1e30f;
      s1[i] = (c1 < T_TOK) ? s1[i] * 0.125f : -1e30f;
    }
    float tm[8], p0[8], p1[8], rs[8], al[8];
#pragma unroll
    for (int i = 0; i < 8; ++i) tm[i] = fmaxf(s0[i], s1[i]);
#pragma unroll
    for (int mk = 1; mk < 16; mk <<= 1) {
#pragma unroll
      for (int i = 0; i < 8; ++i)
        tm[i] = fmaxf(tm[i], __shfl_xor(tm[i], mk, 32));
    }
#pragma unroll
    for (int i = 0; i < 8; ++i) {
      float mn = fmaxf(mrow[i], tm[i]);
      al[i] = __expf(mrow[i] - mn);
      mrow[i] = mn;
      p0[i] = __expf(s0[i] - mn);
      p1[i] = __expf(s1[i] - mn);
      rs[i] = p0[i] + p1[i];
    }
#pragma unroll
    for (int mk = 1; mk < 16; mk <<= 1) {
#pragma unroll
      for (int i = 0; i < 8; ++i) rs[i] += __shfl_xor(rs[i], mk, 32);
    }
#pragma unroll
    for (int i = 0; i < 8; ++i) {
      lrow[i] = lrow[i] * al[i] + rs[i];
      o0[i] *= al[i]; o1[i] *= al[i]; o2[i] *= al[i]; o3[i] *= al[i];
    }
    int hi = lane >> 4;
#pragma unroll
    for (int i = 0; i < 8; ++i) {
      pt[(i + 8 * hi) * 32 + (lane & 15)] = (_Float16)p0[i];
      pt[(i + 8 * hi) * 32 + 16 + (lane & 15)] = (_Float16)p1[i];
    }
    __syncthreads();
    v16h pf = frag_a_lds(pt, 32);
    v16h vf0 = frag_b_kn(vptr, 2304, kt, 0, T_TOK, HDIM);
    v16h vf1 = frag_b_kn(vptr, 2304, kt, 16, T_TOK, HDIM);
    v16h vf2 = frag_b_kn(vptr, 2304, kt, 32, T_TOK, HDIM);
    v16h vf3 = frag_b_kn(vptr, 2304, kt, 48, T_TOK, HDIM);
    o0 = wmma_f32(pf, vf0, o0);
    o1 = wmma_f32(pf, vf1, o1);
    o2 = wmma_f32(pf, vf2, o2);
    o3 = wmma_f32(pf, vf3, o3);
    __syncthreads();
  }
  int n = lane & 15;
  int rb = q0 + ((lane >> 4) << 3);
  float* ob = out + (size_t)b * T_TOK * DMODEL + h * HDIM;
#pragma unroll
  for (int i = 0; i < 8; ++i) {
    int m = rb + i;
    if (m < T_TOK) {
      float inv = 1.f / lrow[i];
      ob[(size_t)m * DMODEL + n]      = o0[i] * inv;
      ob[(size_t)m * DMODEL + 16 + n] = o1[i] * inv;
      ob[(size_t)m * DMODEL + 32 + n] = o2[i] * inv;
      ob[(size_t)m * DMODEL + 48 + n] = o3[i] * inv;
    }
  }
}

// ---------------------------------------------------------------------------
// LayerNorm: one wave per row.
// ---------------------------------------------------------------------------
__global__ __launch_bounds__(32) void ln_kernel(const float* __restrict__ X,
                                                const float* __restrict__ g,
                                                const float* __restrict__ bt,
                                                float* __restrict__ Y, int D) {
  int row = blockIdx.x;
  int lane = threadIdx.x;
  const float* x = X + (size_t)row * D;
  float s = 0.f, sq = 0.f;
  for (int d = lane; d < D; d += 32) {
    float v = x[d];
    s += v;
    sq += v * v;
  }
  for (int mk = 1; mk < 32; mk <<= 1) {
    s += __shfl_xor(s, mk, 32);
    sq += __shfl_xor(sq, mk, 32);
  }
  float mean = s / D;
  float var = sq / D - mean * mean;
  float rs = rsqrtf(var + 1e-6f);
  float* y = Y + (size_t)row * D;
  for (int d = lane; d < D; d += 32) y[d] = (x[d] - mean) * rs * g[d] + bt[d];
}

// Route softmax over E=4 experts; one wave per token.
__global__ __launch_bounds__(256) void route_kernel(
    const float* __restrict__ X, int ld, const float* __restrict__ rW,
    float* __restrict__ route, int Mrow, int din) {
  int t = blockIdx.x * 8 + (threadIdx.x >> 5);
  if (t >= Mrow) return;
  int lane = threadIdx.x & 31;
  const float* x = X + (size_t)t * ld;
  float a0 = 0.f, a1 = 0.f, a2 = 0.f, a3 = 0.f;
  for (int d = lane; d < din; d += 32) {
    float xv = x[d];
    const float* r = rW + (size_t)d * 4;
    a0 += xv * r[0]; a1 += xv * r[1]; a2 += xv * r[2]; a3 += xv * r[3];
  }
  for (int mk = 1; mk < 32; mk <<= 1) {
    a0 += __shfl_xor(a0, mk, 32); a1 += __shfl_xor(a1, mk, 32);
    a2 += __shfl_xor(a2, mk, 32); a3 += __shfl_xor(a3, mk, 32);
  }
  if (lane == 0) {
    float mx = fmaxf(fmaxf(a0, a1), fmaxf(a2, a3));
    float e0 = __expf(a0 - mx), e1 = __expf(a1 - mx);
    float e2 = __expf(a2 - mx), e3 = __expf(a3 - mx);
    float inv = 1.f / (e0 + e1 + e2 + e3);
    route[t * 4 + 0] = e0 * inv; route[t * 4 + 1] = e1 * inv;
    route[t * 4 + 2] = e2 * inv; route[t * 4 + 3] = e3 * inv;
  }
}

// Repack A (E, din, R) -> (din, E*R)
__global__ void pack_kernel(const float* __restrict__ A, float* __restrict__ P,
                            int din, int n) {
  int idx = blockIdx.x * 256 + threadIdx.x;
  if (idx >= n) return;
  int d = idx >> 6, c = idx & 63;
  int e = c >> 4, r = c & 15;
  P[idx] = A[((size_t)e * din + d) * 16 + r];
}

// G[t, e*16+r] *= route[t, e]
__global__ void scaleg_kernel(float* __restrict__ G,
                              const float* __restrict__ route, int n) {
  int idx = blockIdx.x * 256 + threadIdx.x;
  if (idx >= n) return;
  G[idx] *= route[(idx >> 6) * 4 + ((idx & 63) >> 4)];
}

// x += ls[d] * src   (layerscale residual)
__global__ void resid_kernel(float* __restrict__ x,
                             const float* __restrict__ src,
                             const float* __restrict__ ls, int n, int D) {
  int idx = blockIdx.x * 256 + threadIdx.x;
  if (idx >= n) return;
  x[idx] += ls[idx % D] * src[idx];
}

// im2col patch gather: (B,3,h*14,w*14) -> (B, h*w, 588)
__global__ void gather_kernel(const float* __restrict__ img,
                              float* __restrict__ out, int hh, int ww, int n) {
  int idx = blockIdx.x * 256 + threadIdx.x;
  if (idx >= n) return;
  int q = idx % 588;
  int p = (idx / 588) % (hh * ww);
  int b = idx / (588 * hh * ww);
  int c = q / 196, rem = q % 196;
  int i = rem / 14, j = rem % 14;
  int ph = p / ww, pw = p % ww;
  int H = hh * 14, W = ww * 14;
  out[idx] = img[(((size_t)b * 3 + c) * H + ph * 14 + i) * W + pw * 14 + j];
}

// Build fusion token sequence (B, 449, 768)
__global__ void assemble_kernel(const float* __restrict__ zf0,
                                const float* __restrict__ xf0,
                                const float* __restrict__ trackq,
                                const float* __restrict__ qemb,
                                const float* __restrict__ pos,
                                const float* __restrict__ ttype,
                                const int* __restrict__ zmask,
                                float* __restrict__ fusion, int n) {
  int idx = blockIdx.x * 256 + threadIdx.x;
  if (idx >= n) return;
  int d = idx % DMODEL;
  int t = (idx / DMODEL) % T_TOK;
  int b = idx / (DMODEL * T_TOK);
  float v;
  if (t == 0) {
    v = trackq[d] + qemb[d];
  } else if (t < 193) {
    int i = (t - 1) & 63;
    int zh = i >> 3, zw = i & 7;
    v = zf0[((size_t)b * 64 + i) * DMODEL + d] +
        pos[(zh * 16 + zw) * DMODEL + d] +
        ttype[zmask[b * 64 + i] * DMODEL + d];
  } else {
    int j = t - 193;
    v = xf0[((size_t)b * 256 + j) * DMODEL + d] + pos[j * DMODEL + d] +
        ttype[2 * DMODEL + d];
  }
  fusion[idx] = v;
}

// head: att = <enc_opt[t], fusion[0]>, opt = enc_opt * att ; one wave / token
__global__ __launch_bounds__(256) void opt_kernel(const float* __restrict__ fn,
                                                  float* __restrict__ opt) {
  int tg = blockIdx.x * 8 + (threadIdx.x >> 5);
  if (tg >= BATCH * 256) return;
  int b = tg >> 8, t = tg & 255;
  int lane = threadIdx.x & 31;
  const float* fp = fn + ((size_t)b * T_TOK + 193 + t) * DMODEL;
  const float* qp = fn + (size_t)b * T_TOK * DMODEL;
  float acc = 0.f;
  for (int d = lane; d < DMODEL; d += 32) acc += fp[d] * qp[d];
  for (int mk = 1; mk < 32; mk <<= 1) acc += __shfl_xor(acc, mk, 32);
  float* op = opt + (size_t)tg * DMODEL;
  for (int d = lane; d < DMODEL; d += 32) op[d] = fp[d] * acc;
}

// tiny final projection 256 -> 5
__global__ void head2_kernel(const float* __restrict__ hh,
                             const float* __restrict__ W2,
                             const float* __restrict__ b2,
                             float* __restrict__ out, int n) {
  int idx = blockIdx.x * 256 + threadIdx.x;
  if (idx >= n) return;
  int c = idx % 5, row = idx / 5;
  float acc = b2[c];
  const float* hp = hh + (size_t)row * 256;
  for (int k = 0; k < 256; ++k) acc += hp[k] * W2[k * 5 + c];
  out[idx] = acc;
}

// ---------------------------------------------------------------------------
// Host side
// ---------------------------------------------------------------------------
static inline dim3 gemm_grid(int M, int N) {
  return dim3((M + 63) / 64, (N + 63) / 64);
}

static void launch_gemm(hipStream_t s, const float* A1, int lda1,
                        const float* W1, int ldw1, int K1, const float* A2,
                        int lda2, const float* W2, int ldw2, int K2, float s2,
                        const float* bias, float* C, int ldc, int M, int N,
                        int act) {
  gemm_wmma<<<gemm_grid(M, N), 256, 0, s>>>(A1, lda1, W1, ldw1, K1, A2, lda2,
                                            W2, ldw2, K2, s2, bias, C, ldc, M,
                                            N, act);
}

static void run_tmoe(hipStream_t s, const float* X, int din, int dout,
                     const float* W, const float* bias, const float* rW,
                     const float* A, const float* Bm, float* routeb, float* Gb,
                     float* Apk, float* out, int act, int Mrow) {
  route_kernel<<<dim3((Mrow + 7) / 8), 256, 0, s>>>(X, din, rW, routeb, Mrow,
                                                    din);
  int npack = din * LORA_K;
  pack_kernel<<<(npack + 255) / 256, 256, 0, s>>>(A, Apk, din, npack);
  launch_gemm(s, X, din, Apk, LORA_K, din, nullptr, 0, nullptr, 0, 0, 0.f,
              nullptr, Gb, LORA_K, Mrow, LORA_K, 0);
  scaleg_kernel<<<(Mrow * LORA_K + 255) / 256, 256, 0, s>>>(Gb, routeb,
                                                            Mrow * LORA_K);
  launch_gemm(s, X, din, W, dout, din, Gb, LORA_K, Bm, dout, LORA_K,
              SCALE_LORA, bias, out, dout, Mrow, dout, act);
}

extern "C" void kernel_launch(void* const* d_in, const int* in_sizes, int n_in,
                              void* d_out, int out_size, void* d_ws,
                              size_t ws_size, hipStream_t stream) {
  // Inputs in setup_inputs() dict order (params flattened depth-first):
  const float* z0     = (const float*)d_in[1];
  const float* xin    = (const float*)d_in[2];
  const int*   zmask  = (const int*)d_in[3];
  const float* patchW = (const float*)d_in[4];
  const float* patchb = (const float*)d_in[5];
  const float* pos    = (const float*)d_in[6];
  const float* trackq = (const float*)d_in[7];
  const float* qemb   = (const float*)d_in[8];
  const float* ttype  = (const float*)d_in[9];
  const float* g1     = (const float*)d_in[10];
  const float* b1     = (const float*)d_in[11];
  const float* qkvW   = (const float*)d_in[12];
  const float* qkvb   = (const float*)d_in[13];
  const float* qkvrW  = (const float*)d_in[14];
  const float* qkvA   = (const float*)d_in[15];
  const float* qkvBm  = (const float*)d_in[16];
  const float* projW  = (const float*)d_in[17];
  const float* projb  = (const float*)d_in[18];
  const float* projrW = (const float*)d_in[19];
  const float* projA  = (const float*)d_in[20];
  const float* projBm = (const float*)d_in[21];
  const float* ls1    = (const float*)d_in[22];
  const float* g2     = (const float*)d_in[23];
  const float* b2     = (const float*)d_in[24];
  const float* fc1W   = (const float*)d_in[25];
  const float* fc1b   = (const float*)d_in[26];
  const float* fc1rW  = (const float*)d_in[27];
  const float* fc1A   = (const float*)d_in[28];
  const float* fc1Bm  = (const float*)d_in[29];
  const float* fc2W   = (const float*)d_in[30];
  const float* fc2b   = (const float*)d_in[31];
  const float* fc2rW  = (const float*)d_in[32];
  const float* fc2A   = (const float*)d_in[33];
  const float* fc2Bm  = (const float*)d_in[34];
  const float* ls2    = (const float*)d_in[35];
  const float* ng     = (const float*)d_in[36];
  const float* nb     = (const float*)d_in[37];
  const float* hW1    = (const float*)d_in[38];
  const float* hb1    = (const float*)d_in[39];
  const float* hW2    = (const float*)d_in[40];
  const float* hb2    = (const float*)d_in[41];
  float* outp = (float*)d_out;

  // Workspace carve (floats)
  float* w = (float*)d_ws;
  size_t off = 0;
  auto alloc = [&](size_t n) { float* p = w + off; off += n; return p; };
  float* xpz    = alloc((size_t)BATCH * 64 * 588);
  float* xpx    = alloc((size_t)BATCH * 256 * 588);
  float* zf0    = alloc((size_t)BATCH * 64 * DMODEL);
  float* xf0    = alloc((size_t)BATCH * 256 * DMODEL);   // later reused as `opt`
  float* fusion = alloc((size_t)MTOK * DMODEL);
  float* hbuf   = alloc((size_t)MTOK * DMODEL);
  float* routeb = alloc((size_t)MTOK * 4);
  float* Gb     = alloc((size_t)MTOK * LORA_K);
  float* Apk    = alloc((size_t)3072 * LORA_K);
  float* qkvbuf = alloc((size_t)MTOK * 3 * DMODEL);      // later reused as head h
  float* attno  = alloc((size_t)MTOK * DMODEL);
  float* tout   = alloc((size_t)MTOK * 4 * DMODEL);
  (void)ws_size; (void)n_in; (void)in_sizes; (void)out_size;

  // ---- patch embed ----
  int nz = BATCH * 64 * 588;
  gather_kernel<<<(nz + 255) / 256, 256, 0, stream>>>(z0, xpz, 8, 8, nz);
  int nx = BATCH * 256 * 588;
  gather_kernel<<<(nx + 255) / 256, 256, 0, stream>>>(xin, xpx, 16, 16, nx);
  launch_gemm(stream, xpz, 588, patchW, DMODEL, 588, nullptr, 0, nullptr, 0, 0,
              0.f, patchb, zf0, DMODEL, BATCH * 64, DMODEL, 0);
  launch_gemm(stream, xpx, 588, patchW, DMODEL, 588, nullptr, 0, nullptr, 0, 0,
              0.f, patchb, xf0, DMODEL, BATCH * 256, DMODEL, 0);

  int nf = MTOK * DMODEL;
  assemble_kernel<<<(nf + 255) / 256, 256, 0, stream>>>(
      zf0, xf0, trackq, qemb, pos, ttype, zmask, fusion, nf);

  // ---- transformer blocks ----
  for (int l = 0; l < NLAYER; ++l) {
    const float* g1l = g1 + l * DMODEL;
    const float* b1l = b1 + l * DMODEL;
    const float* g2l = g2 + l * DMODEL;
    const float* b2l = b2 + l * DMODEL;
    const float* ls1l = ls1 + l * DMODEL;
    const float* ls2l = ls2 + l * DMODEL;
    const float* qkvWl = qkvW + (size_t)l * DMODEL * 3 * DMODEL;
    const float* qkvbl = qkvb + (size_t)l * 3 * DMODEL;
    const float* qkvrWl = qkvrW + (size_t)l * DMODEL * 4;
    const float* qkvAl = qkvA + (size_t)l * 4 * DMODEL * 16;
    const float* qkvBml = qkvBm + (size_t)l * LORA_K * 3 * DMODEL;
    const float* projWl = projW + (size_t)l * DMODEL * DMODEL;
    const float* projbl = projb + (size_t)l * DMODEL;
    const float* projrWl = projrW + (size_t)l * DMODEL * 4;
    const float* projAl = projA + (size_t)l * 4 * DMODEL * 16;
    const float* projBml = projBm + (size_t)l * LORA_K * DMODEL;
    const float* fc1Wl = fc1W + (size_t)l * DMODEL * 4 * DMODEL;
    const float* fc1bl = fc1b + (size_t)l * 4 * DMODEL;
    const float* fc1rWl = fc1rW + (size_t)l * DMODEL * 4;
    const float* fc1Al = fc1A + (size_t)l * 4 * DMODEL * 16;
    const float* fc1Bml = fc1Bm + (size_t)l * LORA_K * 4 * DMODEL;
    const float* fc2Wl = fc2W + (size_t)l * 4 * DMODEL * DMODEL;
    const float* fc2bl = fc2b + (size_t)l * DMODEL;
    const float* fc2rWl = fc2rW + (size_t)l * 4 * DMODEL * 4;
    const float* fc2Al = fc2A + (size_t)l * 4 * 4 * DMODEL * 16;
    const float* fc2Bml = fc2Bm + (size_t)l * LORA_K * DMODEL;

    // LN1 -> hbuf ; qkv tmoe ; attention
    ln_kernel<<<MTOK, 32, 0, stream>>>(fusion, g1l, b1l, hbuf, DMODEL);
    run_tmoe(stream, hbuf, DMODEL, 3 * DMODEL, qkvWl, qkvbl, qkvrWl, qkvAl,
             qkvBml, routeb, Gb, Apk, qkvbuf, 0, MTOK);
    attn_wmma<<<dim3(BATCH * NHEAD, (T_TOK + 15) / 16), 32, 0, stream>>>(
        qkvbuf, attno);

    // proj tmoe on attention output, layerscale residual
    run_tmoe(stream, attno, DMODEL, DMODEL, projWl, projbl, projrWl, projAl,
             projBml, routeb, Gb, Apk, tout, 0, MTOK);
    resid_kernel<<<(nf + 255) / 256, 256, 0, stream>>>(fusion, tout, ls1l, nf,
                                                       DMODEL);

    // LN2 -> hbuf ; fc1 tmoe (+GELU in epilogue) ; fc2 tmoe ; residual
    ln_kernel<<<MTOK, 32, 0, stream>>>(fusion, g2l, b2l, hbuf, DMODEL);
    run_tmoe(stream, hbuf, DMODEL, 4 * DMODEL, fc1Wl, fc1bl, fc1rWl, fc1Al,
             fc1Bml, routeb, Gb, Apk, tout, 1 /*gelu*/, MTOK);
    run_tmoe(stream, tout, 4 * DMODEL, DMODEL, fc2Wl, fc2bl, fc2rWl, fc2Al,
             fc2Bml, routeb, Gb, Apk, attno, 0, MTOK);
    resid_kernel<<<(nf + 255) / 256, 256, 0, stream>>>(fusion, attno, ls2l, nf,
                                                       DMODEL);
  }

  // ---- head ----
  ln_kernel<<<MTOK, 32, 0, stream>>>(fusion, ng, nb, hbuf, DMODEL);
  float* optb = xf0;  // reuse (exactly B*256*768)
  opt_kernel<<<dim3((BATCH * 256 + 7) / 8), 256, 0, stream>>>(hbuf, optb);
  float* headh = qkvbuf;  // reuse
  launch_gemm(stream, optb, DMODEL, hW1, 256, DMODEL, nullptr, 0, nullptr, 0, 0,
              0.f, hb1, headh, 256, BATCH * 256, 256, 1 /*gelu*/);
  int nout = BATCH * 256 * 5;
  head2_kernel<<<(nout + 255) / 256, 256, 0, stream>>>(headh, hW2, hb2, outp,
                                                       nout);
}
